// FusionDeTrans_7275674600185
// MI455X (gfx1250) — compile-verified, matrix operand order checked by reading, not run on hardware
//
#include <hip/hip_runtime.h>
#include <hip/hip_bf16.h>

#define DEVI __device__ __forceinline__

typedef __attribute__((ext_vector_type(16))) __bf16 bf16x16;
typedef __attribute__((ext_vector_type(8)))  float  f32x8;

// ---- static config (matches reference) ----
static constexpr int Bn  = 4;
static constexpr int BC  = 256, PC = 64, HC = 256;
static constexpr int Him = 200, Wim = 400, Pp = 10;
static constexpr int Hg  = 20, Wg = 40, LQ = 800;
static constexpr int Mtok = Bn * LQ;              // 3200
static constexpr int NH = 8, NP = 8, DH = 32, NL = 2, MLPD = 1024;

// =====================================================================
// CDNA5 async memory->LDS copy (ASYNCcnt-tracked, no VGPR round trip)
// =====================================================================
DEVI void async_copy_b128(const void* gsrc, const void* lds) {
  unsigned ldsoff = (unsigned)(size_t)lds;        // AS(3) offset lives in addr[31:0]
  asm volatile("global_load_async_to_lds_b128 %0, %1, off"
               :: "v"(ldsoff), "v"(gsrc) : "memory");
}
DEVI void async_wait0() {
  asm volatile("s_wait_asynccnt 0x0" ::: "memory");
}

// =====================================================================
// small conversion / layout kernels
// =====================================================================
__global__ void k_cvt_bf16(const float* __restrict__ x, __bf16* __restrict__ y, int n) {
  int i = blockIdx.x * 256 + threadIdx.x;
  if (i < n) y[i] = (__bf16)x[i];
}

// pe weights (HC, Cin, 10, 10) -> bf16 [HC][Cin*128], 100 real + 28 zero pad per channel
__global__ void k_pad_pe(const float* __restrict__ w, __bf16* __restrict__ dst, int cin) {
  int i = blockIdx.x * 256 + threadIdx.x;
  int total = HC * cin * 128;
  if (i >= total) return;
  int p = i & 127;
  int c = (i >> 7) % cin;
  int n = i / (cin * 128);
  dst[i] = (p < 100) ? (__bf16)w[(n * cin + c) * 100 + p] : (__bf16)0.0f;
}

// expand_w (P*P*BC, HC) row n = py*2560+px*256+cc  ->  bf16 rows ordered n' = cc*100+py*10+px
__global__ void k_reorder_expand(const float* __restrict__ w, __bf16* __restrict__ dst) {
  int i = blockIdx.x * 256 + threadIdx.x;          // 25600*256 total
  if (i >= 25600 * 256) return;
  int k  = i & 255;
  int np = i >> 8;
  int cc = np / 100, p = np % 100, py = p / 10, px = p - py * 10;
  int srow = (py * Pp + px) * BC + cc;
  dst[i] = (__bf16)w[(size_t)srow * HC + k];
}

// =====================================================================
// WMMA helpers — per-lane fragment loads per ISA §7.12.2
// A(16x32)/B(32x16) bf16: lane owns two 8-elem chunks at k=half*8 and 16+half*8
// =====================================================================
DEVI bf16x16 load_frag(const __bf16* base, int half) {
  union { bf16x16 v; uint4 q[2]; } u;
  u.q[0] = *(const uint4*)(base + half * 8);
  u.q[1] = *(const uint4*)(base + 16 + half * 8);
  return u.v;
}

// =====================================================================
// PatchEmbed: strided conv as GEMM with fused im2col gather (f32->bf16 on the fly)
// out[3200][256] = patches(bf16, padded K=Cin*128) @ Wpe^T + bias
// =====================================================================
template <int CIN>
__global__ void k_patch_embed(const float* __restrict__ src, const __bf16* __restrict__ wpe,
                              const float* __restrict__ bias, float* __restrict__ outRaw) {
  __shared__ __align__(32) __bf16 Ps[32 * 128];     // 32 tokens x 128 (padded) bf16
  const int tid  = threadIdx.x;
  const int wave = tid >> 5, lane = tid & 31;
  const int wM = wave & 1, wN = wave >> 1;
  const int mBase = blockIdx.x * 32;
  const int nBase = blockIdx.y * 64;
  const int Kpad  = CIN * 128;

  // staging coords: 8 threads per token
  const int tk = tid >> 3, l8 = tid & 7;
  const int tok  = mBase + tk;
  const int b    = tok / LQ, cell = tok % LQ;
  const int gy   = cell / Wg, gx = cell % Wg;
  const float* sbase = src + (size_t)b * CIN * Him * Wim + (size_t)(gy * Pp) * Wim + gx * Pp;

  const int mRow = wM * 16 + (lane & 15);
  const int half = lane >> 4;
  const int nCol = nBase + wN * 16 + (lane & 15);
  const __bf16* wrow = wpe + (size_t)nCol * Kpad;

  f32x8 acc = {};
  for (int c = 0; c < CIN; ++c) {
    const float* sc = sbase + (size_t)c * Him * Wim;
    __builtin_prefetch(sc + Him * Wim, 0, 1);       // next channel -> global_prefetch_b8
#pragma unroll
    for (int j = 0; j < 16; ++j) {
      int col = l8 + j * 8;
      float v = 0.f;
      if (col < 100) { int py = col / 10, px = col - py * 10; v = sc[py * Wim + px]; }
      Ps[tk * 128 + col] = (__bf16)v;
    }
    __syncthreads();
    const __bf16* wc = wrow + c * 128;
#pragma unroll
    for (int kk = 0; kk < 128; kk += 32) {
      bf16x16 a = load_frag(&Ps[mRow * 128 + kk], half);
      bf16x16 bm = load_frag(wc + kk, half);
      acc = __builtin_amdgcn_wmma_f32_16x16x32_bf16(false, a, false, bm, (short)0, acc, false, false);
    }
    __syncthreads();
  }
  float bv = bias[nCol];
#pragma unroll
  for (int r = 0; r < 8; ++r) {
    int m = mBase + wM * 16 + r + (half << 3);
    outRaw[(size_t)m * HC + nCol] = acc[r] + bv;
  }
}

// =====================================================================
// Generic bf16 GEMM: C[M][N] = A[M][K] @ W[N][K]^T (+bias), K % 128 == 0.
// A staged to LDS in 32x128 chunks via global_load_async_to_lds_b128;
// 4 WMMAs per barrier round. B streamed from global (L2-resident weights).
// MODE 0: store f32   MODE 1: += into f32 (residual)
// MODE 2: GELU -> bf16   MODE 3: expand scatter + bev residual -> d_out
// =====================================================================
template <int MODE>
__global__ void k_gemm(const __bf16* __restrict__ A, const __bf16* __restrict__ W,
                       const float* __restrict__ bias, int M, int N, int K,
                       float* __restrict__ out0, const float* __restrict__ aux,
                       __bf16* __restrict__ outB) {
  __shared__ __align__(32) __bf16 As[32 * 128];     // 8 KB
  const int tid  = threadIdx.x;
  const int wave = tid >> 5, lane = tid & 31;
  const int wM = wave & 1, wN = wave >> 1;
  const int mBase = blockIdx.x * 32, nBase = blockIdx.y * 64;

  const int mRow = wM * 16 + (lane & 15);
  const int half = lane >> 4;
  const int nCol = nBase + wN * 16 + (lane & 15);
  const __bf16* wrow = W + (size_t)nCol * K;

  // staging: each thread owns 32 bytes (16 bf16) of the 32x128 tile
  const int arow = tid >> 3;
  const int acol = (tid & 7) * 16;
  const __bf16* aptr = A + (size_t)(mBase + arow) * K + acol;
  const __bf16* lptr = &As[arow * 128 + acol];

  f32x8 acc = {};
  for (int k0 = 0; k0 < K; k0 += 128) {
    async_copy_b128(aptr + k0,     lptr);
    async_copy_b128(aptr + k0 + 8, lptr + 8);
    __builtin_prefetch(aptr + k0 + 128, 0, 1);
    __builtin_prefetch(wrow + k0 + 128, 0, 1);
    async_wait0();
    __syncthreads();
#pragma unroll
    for (int kk = 0; kk < 128; kk += 32) {
      bf16x16 a = load_frag(&As[mRow * 128 + kk], half);
      bf16x16 bm = load_frag(wrow + k0 + kk, half);
      acc = __builtin_amdgcn_wmma_f32_16x16x32_bf16(false, a, false, bm, (short)0, acc, false, false);
    }
    __syncthreads();
  }

  float bval = bias ? bias[nCol] : 0.f;
#pragma unroll
  for (int r = 0; r < 8; ++r) {
    int m = mBase + wM * 16 + r + (half << 3);
    float v = acc[r] + bval;
    if constexpr (MODE == 0) {
      out0[(size_t)m * N + nCol] = v;
    } else if constexpr (MODE == 1) {
      out0[(size_t)m * N + nCol] += v;
    } else if constexpr (MODE == 2) {
      float g = 0.5f * v * (1.0f + erff(v * 0.70710678118f));   // exact GELU
      outB[(size_t)m * N + nCol] = (__bf16)g;
    } else {                                                    // expand scatter
      int b = m / LQ, cell = m % LQ, gy = cell / Wg, gx = cell % Wg;
      int cc = nCol / 100, p = nCol % 100, py = p / 10, px = p - py * 10;
      size_t oi = (((size_t)b * BC + cc) * Him + gy * Pp + py) * (size_t)Wim + gx * Pp + px;
      out0[oi] = v + aux[oi];
    }
  }
}

// =====================================================================
// LayerNorm over HC=256 (one block per row), optional +pos, f32/bf16 outs.
// In-place safe (each element read by its own thread before any write).
// =====================================================================
__global__ void k_layernorm(const float* __restrict__ x, const float* __restrict__ pos,
                            const float* __restrict__ g, const float* __restrict__ bb,
                            float* __restrict__ outF, __bf16* __restrict__ outB) {
  __shared__ float red[256];
  int r = blockIdx.x, c = threadIdx.x;
  float v = x[(size_t)r * HC + c];
  if (pos) v += pos[(size_t)(r % LQ) * HC + c];
  red[c] = v; __syncthreads();
  for (int s = 128; s > 0; s >>= 1) { if (c < s) red[c] += red[c + s]; __syncthreads(); }
  float mean = red[0] * (1.f / HC); __syncthreads();
  float d = v - mean; red[c] = d * d; __syncthreads();
  for (int s = 128; s > 0; s >>= 1) { if (c < s) red[c] += red[c + s]; __syncthreads(); }
  float rstd = rsqrtf(red[0] * (1.f / HC) + 1e-5f);
  float o = d * rstd * g[c] + bb[c];
  if (outF) outF[(size_t)r * HC + c] = o;
  if (outB) outB[(size_t)r * HC + c] = (__bf16)o;
}

// softmax over NP=8 per (token, head); aw layout [M][NH*NP], in place
__global__ void k_softmax8(float* __restrict__ aw) {
  int i = blockIdx.x * 256 + threadIdx.x;           // (t*NH + head)
  if (i >= Mtok * NH) return;
  float* p = aw + (size_t)(i >> 3) * (NH * NP) + (i & 7) * NP;
  float mx = p[0];
#pragma unroll
  for (int j = 1; j < 8; ++j) mx = fmaxf(mx, p[j]);
  float e[8], s = 0.f;
#pragma unroll
  for (int j = 0; j < 8; ++j) { e[j] = expf(p[j] - mx); s += e[j]; }
  float inv = 1.f / s;
#pragma unroll
  for (int j = 0; j < 8; ++j) p[j] = e[j] * inv;
}

// deformable bilinear sampling + head-weighted sum. One wave per (token, head),
// lane = channel d (0..31). loc simplifies to x = gx + off_x, y = gy + off_y.
__global__ void k_deform(const float* __restrict__ value, const float* __restrict__ off,
                         const float* __restrict__ aw, __bf16* __restrict__ attnB) {
  int wave = threadIdx.x >> 5, lane = threadIdx.x & 31;
  int uid = blockIdx.x * 8 + wave;                  // 25600 units
  int t = uid >> 3, head = uid & 7;
  int b = t / LQ, cell = t % LQ, gy = cell / Wg, gx = cell % Wg;
  const float* vb = value + (size_t)b * LQ * HC + head * DH + lane;
  float acc = 0.f;
#pragma unroll
  for (int p = 0; p < NP; ++p) {
    float ox = off[(size_t)t * (NH * NP * 2) + (head * NP + p) * 2 + 0];
    float oy = off[(size_t)t * (NH * NP * 2) + (head * NP + p) * 2 + 1];
    float wv = aw[(size_t)t * (NH * NP) + head * NP + p];
    float x = (float)gx + ox, y = (float)gy + oy;
    float xf = floorf(x), yf = floorf(y);
    int x0 = (int)xf, y0 = (int)yf;
    float lx = x - xf, ly = y - yf;
    float s = 0.f;
#pragma unroll
    for (int cy = 0; cy < 2; ++cy)
#pragma unroll
      for (int cx = 0; cx < 2; ++cx) {
        int xi = x0 + cx, yi = y0 + cy;
        float w2 = (cx ? lx : 1.f - lx) * (cy ? ly : 1.f - ly);
        if (xi >= 0 && xi < Wg && yi >= 0 && yi < Hg)
          s += w2 * vb[(size_t)(yi * Wg + xi) * HC];
      }
    acc += wv * s;
  }
  attnB[(size_t)t * HC + head * DH + lane] = (__bf16)acc;
}

// =====================================================================
// host orchestration
// =====================================================================
extern "C" void kernel_launch(void* const* d_in, const int* in_sizes, int n_in,
                              void* d_out, int out_size, void* d_ws, size_t ws_size,
                              hipStream_t stream) {
  const float* bev       = (const float*)d_in[0];
  const float* prior     = (const float*)d_in[1];
  const float* pe_bev_w  = (const float*)d_in[2];
  const float* pe_bev_b  = (const float*)d_in[3];
  const float* pe_pr_w   = (const float*)d_in[4];
  const float* pe_pr_b   = (const float*)d_in[5];
  const float* pos_bev   = (const float*)d_in[6];
  const float* pos_pr    = (const float*)d_in[7];
  const float* ln_bev_g  = (const float*)d_in[8];
  const float* ln_bev_b  = (const float*)d_in[9];
  const float* ln_pr_g   = (const float*)d_in[10];
  const float* ln_pr_b   = (const float*)d_in[11];
  const float* n1_g      = (const float*)d_in[12];
  const float* n1_b      = (const float*)d_in[13];
  const float* n2_g      = (const float*)d_in[14];
  const float* n2_b      = (const float*)d_in[15];
  const float* off_w     = (const float*)d_in[16];
  const float* off_b     = (const float*)d_in[17];
  const float* aw_w      = (const float*)d_in[18];
  const float* aw_b      = (const float*)d_in[19];
  const float* vp_w      = (const float*)d_in[20];
  const float* vp_b      = (const float*)d_in[21];
  const float* op_w      = (const float*)d_in[22];
  const float* op_b      = (const float*)d_in[23];
  const float* fc1_w     = (const float*)d_in[24];
  const float* fc1_b     = (const float*)d_in[25];
  const float* fc2_w     = (const float*)d_in[26];
  const float* fc2_b     = (const float*)d_in[27];
  const float* expand_w  = (const float*)d_in[28];
  float* out = (float*)d_out;

  char* ws = (char*)d_ws;
  size_t cur = 0;
  auto alloc = [&](size_t bytes) -> void* {
    void* p = ws + cur;
    cur = (cur + bytes + 255) & ~(size_t)255;
    return p;
  };

  __bf16* wPeBev = (__bf16*)alloc((size_t)HC * BC * 128 * 2);       // 16.8 MB
  __bf16* wPePr  = (__bf16*)alloc((size_t)HC * PC * 128 * 2);       //  4.2 MB
  __bf16* wExp   = (__bf16*)alloc((size_t)25600 * HC * 2);          // 13.1 MB
  __bf16* wVp    = (__bf16*)alloc((size_t)NL * HC * HC * 2);
  __bf16* wOff   = (__bf16*)alloc((size_t)NL * 128 * HC * 2);
  __bf16* wAw    = (__bf16*)alloc((size_t)NL * 64 * HC * 2);
  __bf16* wOp    = (__bf16*)alloc((size_t)NL * HC * HC * 2);
  __bf16* wFc1   = (__bf16*)alloc((size_t)NL * MLPD * HC * 2);
  __bf16* wFc2   = (__bf16*)alloc((size_t)NL * HC * MLPD * 2);
  float*  tgt    = (float*)alloc((size_t)Mtok * HC * 4);
  float*  pr     = (float*)alloc((size_t)Mtok * HC * 4);
  __bf16* tlnB   = (__bf16*)alloc((size_t)Mtok * HC * 2);
  __bf16* slnB   = (__bf16*)alloc((size_t)Mtok * HC * 2);
  float*  valueF = (float*)alloc((size_t)Mtok * HC * 4);
  float*  offF   = (float*)alloc((size_t)Mtok * 128 * 4);
  float*  awF    = (float*)alloc((size_t)Mtok * 64 * 4);
  __bf16* attnB  = (__bf16*)alloc((size_t)Mtok * HC * 2);
  __bf16* hB     = (__bf16*)alloc((size_t)Mtok * MLPD * 2);
  __bf16* tgtB   = (__bf16*)alloc((size_t)Mtok * HC * 2);
  (void)ws_size; (void)n_in; (void)in_sizes; (void)out_size;

  // ---- weight conversion ----
  k_pad_pe<<<(HC * BC * 128) / 256, 256, 0, stream>>>(pe_bev_w, wPeBev, BC);
  k_pad_pe<<<(HC * PC * 128) / 256, 256, 0, stream>>>(pe_pr_w, wPePr, PC);
  k_reorder_expand<<<(25600 * 256) / 256, 256, 0, stream>>>(expand_w, wExp);
  k_cvt_bf16<<<(NL * HC * HC) / 256, 256, 0, stream>>>(vp_w, wVp, NL * HC * HC);
  k_cvt_bf16<<<(NL * 128 * HC) / 256, 256, 0, stream>>>(off_w, wOff, NL * 128 * HC);
  k_cvt_bf16<<<(NL * 64 * HC) / 256, 256, 0, stream>>>(aw_w, wAw, NL * 64 * HC);
  k_cvt_bf16<<<(NL * HC * HC) / 256, 256, 0, stream>>>(op_w, wOp, NL * HC * HC);
  k_cvt_bf16<<<(NL * MLPD * HC) / 256, 256, 0, stream>>>(fc1_w, wFc1, NL * MLPD * HC);
  k_cvt_bf16<<<(NL * HC * MLPD) / 256, 256, 0, stream>>>(fc2_w, wFc2, NL * HC * MLPD);

  // ---- patch embed + pos + LN ----
  dim3 gPE(Mtok / 32, HC / 64);
  k_patch_embed<BC><<<gPE, 256, 0, stream>>>(bev, wPeBev, pe_bev_b, tgt);
  k_patch_embed<PC><<<gPE, 256, 0, stream>>>(prior, wPePr, pe_pr_b, pr);
  k_layernorm<<<Mtok, 256, 0, stream>>>(tgt, pos_bev, ln_bev_g, ln_bev_b, tgt, nullptr);
  k_layernorm<<<Mtok, 256, 0, stream>>>(pr, pos_pr, ln_pr_g, ln_pr_b, pr, nullptr);

  // ---- decoder layers ----
  for (int l = 0; l < NL; ++l) {
    const float* g1 = n1_g + l * HC; const float* b1 = n1_b + l * HC;
    const float* g2 = n2_g + l * HC; const float* b2 = n2_b + l * HC;
    k_layernorm<<<Mtok, 256, 0, stream>>>(tgt, nullptr, g1, b1, nullptr, tlnB);
    k_layernorm<<<Mtok, 256, 0, stream>>>(pr, nullptr, g1, b1, nullptr, slnB);

    k_gemm<0><<<dim3(Mtok / 32, HC / 64), 256, 0, stream>>>(
        slnB, wVp + (size_t)l * HC * HC, vp_b + l * HC, Mtok, HC, HC, valueF, nullptr, nullptr);
    k_gemm<0><<<dim3(Mtok / 32, 128 / 64), 256, 0, stream>>>(
        tlnB, wOff + (size_t)l * 128 * HC, off_b + l * 128, Mtok, 128, HC, offF, nullptr, nullptr);
    k_gemm<0><<<dim3(Mtok / 32, 64 / 64), 256, 0, stream>>>(
        tlnB, wAw + (size_t)l * 64 * HC, aw_b + l * 64, Mtok, 64, HC, awF, nullptr, nullptr);
    k_softmax8<<<(Mtok * NH + 255) / 256, 256, 0, stream>>>(awF);
    k_deform<<<Mtok, 256, 0, stream>>>(valueF, offF, awF, attnB);
    k_gemm<1><<<dim3(Mtok / 32, HC / 64), 256, 0, stream>>>(
        attnB, wOp + (size_t)l * HC * HC, op_b + l * HC, Mtok, HC, HC, tgt, nullptr, nullptr);

    k_layernorm<<<Mtok, 256, 0, stream>>>(tgt, nullptr, g2, b2, nullptr, tlnB);
    k_gemm<2><<<dim3(Mtok / 32, MLPD / 64), 256, 0, stream>>>(
        tlnB, wFc1 + (size_t)l * MLPD * HC, fc1_b + l * MLPD, Mtok, MLPD, HC, nullptr, nullptr, hB);
    k_gemm<1><<<dim3(Mtok / 32, HC / 64), 256, 0, stream>>>(
        hB, wFc2 + (size_t)l * HC * MLPD, fc2_b + l * HC, Mtok, HC, MLPD, tgt, nullptr, nullptr);
  }

  // ---- expand + residual into d_out ----
  k_cvt_bf16<<<(Mtok * HC) / 256, 256, 0, stream>>>(tgt, tgtB, Mtok * HC);
  k_gemm<3><<<dim3(Mtok / 32, 25600 / 64), 256, 0, stream>>>(
      tgtB, wExp, nullptr, Mtok, 25600, HC, out, bev, nullptr);
}
